// MemoryAttention_90872918049500
// MI455X (gfx1250) — matrix operand order repaired and checked by reading, not compiled, not run on hardware
//
#include <hip/hip_runtime.h>

typedef __attribute__((ext_vector_type(16))) __bf16 v16bf;
typedef __attribute__((ext_vector_type(8)))  float  v8f;
typedef __attribute__((ext_vector_type(4))) unsigned int u32x4;
typedef __attribute__((ext_vector_type(8))) int          i32x8;
typedef __attribute__((ext_vector_type(4))) int          i32x4;

using u32 = unsigned int;
using u16 = unsigned short;
using u64 = unsigned long long;

#define B_  8
#define T_  512
#define S_  2048
#define D_  1024
#define H_  256
#define NH_ 4
#define MASK_FILL (-2.3819763e+38f)

#if __has_builtin(__builtin_amdgcn_tensor_load_to_lds) && \
    __has_builtin(__builtin_amdgcn_s_wait_tensorcnt)
#define HAS_TDM 1
#else
#define HAS_TDM 0
#endif

__device__ __forceinline__ u16 f2bf(float f) {
  u32 u = __builtin_bit_cast(u32, f);
  u32 r = (u + 0x7FFFu + ((u >> 16) & 1u)) >> 16;
  return (u16)r;
}
__device__ __forceinline__ float bf2f(u16 h) {
  u32 u = ((u32)h) << 16;
  return __builtin_bit_cast(float, u);
}

struct Frag { union { v16bf v; u32 u[8]; }; };

// A fragment (16x32 bf16, row-major source). rowPairs = &A[m][k0] as dword pairs.
// ISA 7.12.2: lane<16 holds K 0..7 (v0..3) and 16..23 (v4..7); lane>=16 holds 8..15 / 24..31.
__device__ __forceinline__ void loadA(Frag& f, const u32* rowPairs, int half) {
#pragma unroll
  for (int i = 0; i < 4; ++i) {
    f.u[i]     = rowPairs[half * 4 + i];
    f.u[4 + i] = rowPairs[8 + half * 4 + i];
  }
}
// B fragment from N-major (Bt is N x K row-major). rowPairs = &Bt[n][k0].
__device__ __forceinline__ void loadB(Frag& f, const u32* rowPairs, int half) {
#pragma unroll
  for (int i = 0; i < 4; ++i) {
    f.u[i]     = rowPairs[half * 4 + i];
    f.u[4 + i] = rowPairs[8 + half * 4 + i];
  }
}

#if HAS_TDM
// TDM: DMA one 32(s) x 256(h) u16 tile, row-major, global -> LDS.
// D# group0: count=1 | lds_addr | global_addr(57b) | type=2 ("image")
// D# group1: data_size=1(2B) | tensor_dim0=256 | tensor_dim1=32 | tile_dim0=256
//            | tile_dim1=32 | tensor_dim0_stride=256
__device__ __forceinline__ void tdm_load_k_tile(u32 lds_addr, const u16* gptr) {
  u64 ga = (u64)(size_t)gptr;
  u32x4 g0;
  g0[0] = 1u;                             // count=1, user descriptor
  g0[1] = lds_addr;                       // LDS byte address (bits 63:32)
  g0[2] = (u32)ga;                        // global addr lo
  g0[3] = (u32)(ga >> 32) | (2u << 30);   // global addr hi | type=2
  i32x8 g1;
  g1[0] = (int)(1u << 16);                // wg_mask=0, data_size=1 (2 bytes)
  g1[1] = (int)(256u << 16);              // tensor_dim0[15:0]=256 -> bits 63:48
  g1[2] = (int)(32u << 16);               // tensor_dim1[15:0]=32  -> bits 95:80
  g1[3] = (int)(256u << 16);              // tile_dim0=256         -> bits 127:112
  g1[4] = 32;                             // tile_dim1=32, tile_dim2=0
  g1[5] = 256;                            // tensor_dim0_stride lo32 = 256
  g1[6] = 0;                              // stride0 hi / stride1 lo
  g1[7] = 0;
  i32x4 gz4 = {0, 0, 0, 0};
  i32x8 gz8 = {0, 0, 0, 0, 0, 0, 0, 0};
  // 6-arg form (clang-23 / therock-10.0 headers): groups 0..3 + extra group + cpol
  __builtin_amdgcn_tensor_load_to_lds(g0, g1, gz4, gz4, gz8, 0);
}
#endif

// ---------------- RMSNorm -> bf16 ----------------
__global__ void __launch_bounds__(256) rmsnorm_bf16(const float* __restrict__ x,
                                                    const float* __restrict__ scale,
                                                    u16* __restrict__ out) {
  int row = blockIdx.x;
  int tid = threadIdx.x;
  const float* xr = x + (size_t)row * D_;
  float ss = 0.f;
#pragma unroll
  for (int i = 0; i < 4; ++i) { float v = xr[tid + i * 256]; ss += v * v; }
#pragma unroll
  for (int off = 1; off < 32; off <<= 1) ss += __shfl_xor(ss, off, 32);
  __shared__ float part[8];
  if ((tid & 31) == 0) part[tid >> 5] = ss;
  __syncthreads();
  float tot = 0.f;
#pragma unroll
  for (int i = 0; i < 8; ++i) tot += part[i];
  float inv = rsqrtf(tot * (1.0f / D_) + 1e-6f);
  u16* orow = out + (size_t)row * D_;
#pragma unroll
  for (int i = 0; i < 4; ++i) {
    int c = tid + i * 256;
    orow[c] = f2bf(xr[c] * inv * (1.0f + scale[c]));
  }
}

// ---------------- weight conversions (f32 -> bf16, pre-transposed to NxK) ------
__global__ void conv_qw(const float* __restrict__ qw, u16* __restrict__ qwt) {
  int id = blockIdx.x * 256 + threadIdx.x;  // 1M
  int c = id >> 10, d = id & 1023;
  int n = c >> 8, h = c & 255;
  qwt[(size_t)c * 1024 + d] = f2bf(qw[(size_t)n * 262144 + d * 256 + h]);
}
__global__ void conv_kvw(const float* __restrict__ kvw, u16* __restrict__ kwt,
                         u16* __restrict__ vwt) {
  int id = blockIdx.x * 256 + threadIdx.x;  // 262144
  int h = id >> 10, d = id & 1023;
  kwt[(size_t)h * 1024 + d] = f2bf(kvw[(size_t)d * 256 + h]);
  vwt[(size_t)h * 1024 + d] = f2bf(kvw[262144 + (size_t)d * 256 + h]);
}
__global__ void conv_ow(const float* __restrict__ ow, u16* __restrict__ owt) {
  int id = blockIdx.x * 256 + threadIdx.x;
  int d = id >> 10, k = id & 1023;
  owt[(size_t)d * 1024 + k] = f2bf(ow[(size_t)k * 1024 + d]);
}

// ---------------- generic bf16 WMMA GEMM: C = A(MxK) * Bt(NxK)^T --------------
template <bool BF16OUT>
__global__ void __launch_bounds__(256) gemm_wmma(const u16* __restrict__ A,
                                                 const u16* __restrict__ Bt,
                                                 void* __restrict__ Cv,
                                                 int M, int N, int K) {
  int lane = threadIdx.x & 31;
  int wid  = threadIdx.x >> 5;
  int tilesN = N >> 4;
  int tile = blockIdx.x * 8 + wid;
  if (tile >= (M >> 4) * tilesN) return;  // wave-uniform: EXEC stays full for WMMA
  int tm = tile / tilesN, tn = tile - tm * tilesN;
  int m = lane & 15, half = lane >> 4;
  const u32* arow = (const u32*)(A  + (size_t)(tm * 16 + m) * K);
  const u32* brow = (const u32*)(Bt + (size_t)(tn * 16 + m) * K);
  v8f c = {};
  for (int k0 = 0; k0 < K; k0 += 32) {
    Frag a, b;
    int p = k0 >> 1;
    loadA(a, arow + p, half);
    loadB(b, brow + p, half);
    __builtin_prefetch(arow + p + 64, 0, 1);  // global_prefetch_b8
    c = __builtin_amdgcn_wmma_f32_16x16x32_bf16(false, a.v, false, b.v,
                                                (short)0, c, false, false);
  }
  size_t rb = (size_t)(tm * 16 + half * 8);
  int col = tn * 16 + m;
  if (BF16OUT) {
    u16* C = (u16*)Cv;
#pragma unroll
    for (int r = 0; r < 8; ++r) C[(rb + r) * N + col] = f2bf(c[r]);
  } else {
    float* C = (float*)Cv;
#pragma unroll
    for (int r = 0; r < 8; ++r) C[(rb + r) * N + col] = c[r];
  }
}

// ---------------- RoPE (in place on bf16, optional 1/sqrt(H) scale) -----------
__global__ void rope_bf16(u16* __restrict__ q, int nheads, int tmask, int posoff,
                          float oscale, int total) {
  int id = blockIdx.x * 256 + threadIdx.x;
  if (id >= total) return;
  int i = id & 127;
  int t2 = id >> 7;
  int hn = t2 % nheads;
  int row = t2 / nheads;
  int pos = (row & tmask) + posoff;
  size_t base = ((size_t)row * nheads + hn) * 256;
  float x1 = bf2f(q[base + i]);
  float x2 = bf2f(q[base + 128 + i]);
  float inv_ts = __powf(10000.0f, -(float)i * (2.0f / 256.0f));
  float ang = (float)pos * inv_ts;
  float sn = __sinf(ang), cs = __cosf(ang);
  q[base + i]       = f2bf((x1 * cs - x2 * sn) * oscale);
  q[base + 128 + i] = f2bf((x2 * cs + x1 * sn) * oscale);
}

// ---------------- fused flash attention, double-buffered + TDM ----------------
// grid = B*NH*(T/128); block = 256 (8 waves). Each wave owns a 16-row T tile.
// K tile (32x256) is DMA'd by the Tensor Data Mover into LDS (TENSORcnt);
// V tile is cooperatively transposed into LDS. Both double-buffered so the
// next chunk's data movement overlaps this chunk's WMMA work.
__global__ void __launch_bounds__(256) attn_flash(const u16* __restrict__ q,
                                                  const u16* __restrict__ k,
                                                  const u16* __restrict__ v,
                                                  const unsigned char* __restrict__ mask,
                                                  u16* __restrict__ enc) {
  __shared__ u16 kT[2][32 * 256];     // 2 x 16 KB: [s_local][h]
  __shared__ u16 vT[2][256 * 32];     // 2 x 16 KB: [h][s_local]
  __shared__ u16 probs[8][16 * 32];   //     8 KB: per-wave P tile (C->A transpose)

  int tid = threadIdx.x;
  int lane = tid & 31, wid = tid >> 5;
  int m = lane & 15, half = lane >> 4;

  int bid  = blockIdx.x;
  int tb   = bid & 3;
  int head = (bid >> 2) & 3;
  int b    = bid >> 4;

  int rowbase = b * T_ + tb * 128 + wid * 16;
  const u32* qpairs = (const u32*)(q + ((size_t)(rowbase + m)) * D_ + head * H_);
  const u16* kg = k + (size_t)b * S_ * H_;
  const u16* vg = v + (size_t)b * S_ * H_;
  const unsigned char* mb = mask + (size_t)b * S_;

  v8f zero = {};
  v8f acc[16];
#pragma unroll
  for (int i = 0; i < 16; ++i) acc[i] = zero;
  float rowmax[8], rowsum[8];
#pragma unroll
  for (int r = 0; r < 8; ++r) { rowmax[r] = -3.0e38f; rowsum[r] = 0.f; }

  auto stageV = [&](int buf, int s0) {
    for (int e = tid; e < 32 * 256; e += 256) {
      int s = e >> 8, h = e & 255;
      vT[buf][h * 32 + s] = vg[(size_t)(s0 + s) * H_ + h];
    }
  };
#if !HAS_TDM
  auto stageK = [&](int buf, int s0) {
    for (int e = tid; e < 32 * 256; e += 256) {
      int s = e >> 8, h = e & 255;
      kT[buf][e] = kg[(size_t)(s0 + s) * H_ + h];
    }
  };
#endif

  // prologue: fill buffer 0
#if HAS_TDM
  if (wid == 0) tdm_load_k_tile((u32)(size_t)&kT[0][0], kg);
#else
  stageK(0, 0);
#endif
  stageV(0, 0);
#if HAS_TDM
  if (wid == 0) __builtin_amdgcn_s_wait_tensorcnt(0);
#endif
  __syncthreads();

  const int NCH = S_ / 32;
  for (int ic = 0; ic < NCH; ++ic) {
    int cur = ic & 1, nxt = cur ^ 1;
    int s0 = ic * 32;

    // kick off data movement for the next chunk (overlaps compute below)
    if (ic + 1 < NCH) {
#if HAS_TDM
      if (wid == 0)
        tdm_load_k_tile((u32)(size_t)&kT[nxt][0], kg + (size_t)(s0 + 32) * H_);
#else
      stageK(nxt, s0 + 32);
#endif
      stageV(nxt, s0 + 32);
    }

    // logits: 16x32 = q(16x256) . k^T(256x32), two 16x16 WMMA tiles sharing A
    v8f c0 = zero, c1 = zero;
    const u32* b0p = (const u32*)(&kT[cur][m * 256]);
    const u32* b1p = (const u32*)(&kT[cur][(m + 16) * 256]);
#pragma unroll
    for (int h0 = 0; h0 < H_; h0 += 32) {
      Frag a, b0, b1;
      loadA(a,  qpairs + (h0 >> 1), half);
      loadB(b0, b0p    + (h0 >> 1), half);
      loadB(b1, b1p    + (h0 >> 1), half);
      c0 = __builtin_amdgcn_wmma_f32_16x16x32_bf16(false, a.v, false, b0.v,
                                                   (short)0, c0, false, false);
      c1 = __builtin_amdgcn_wmma_f32_16x16x32_bf16(false, a.v, false, b1.v,
                                                   (short)0, c1, false, false);
    }

    // mask (per-lane column)
    bool mk0 = mb[s0 + m] != 0;
    bool mk1 = mb[s0 + 16 + m] != 0;
#pragma unroll
    for (int r = 0; r < 8; ++r) {
      c0[r] = mk0 ? c0[r] : MASK_FILL;
      c1[r] = mk1 ? c1[r] : MASK_FILL;
    }

    // online softmax (row = half*8 + r; cols across the 16-lane half-group)
    float scal[8];
#pragma unroll
    for (int r = 0; r < 8; ++r) {
      float mx = fmaxf(c0[r], c1[r]);
#pragma unroll
      for (int off = 1; off < 16; off <<= 1) mx = fmaxf(mx, __shfl_xor(mx, off, 32));
      float mnew = fmaxf(rowmax[r], mx);
      float sc = __expf(rowmax[r] - mnew);
      rowmax[r] = mnew;
      float p0 = __expf(c0[r] - mnew);
      float p1 = __expf(c1[r] - mnew);
      float ps = p0 + p1;
#pragma unroll
      for (int off = 1; off < 16; off <<= 1) ps += __shfl_xor(ps, off, 32);
      rowsum[r] = rowsum[r] * sc + ps;
      scal[r] = sc;
      probs[wid][(half * 8 + r) * 32 + m]      = f2bf(p0);
      probs[wid][(half * 8 + r) * 32 + 16 + m] = f2bf(p1);
    }
#pragma unroll
    for (int i = 0; i < 16; ++i)
#pragma unroll
      for (int r = 0; r < 8; ++r) acc[i][r] *= scal[r];

    __syncthreads();  // probs C->A transpose visible across lanes

    // acc += P(16x32) . V(32x256)
    Frag ap;
    loadA(ap, (const u32*)(&probs[wid][m * 32]), half);
#pragma unroll
    for (int i = 0; i < 16; ++i) {
      Frag bv;
      loadB(bv, (const u32*)(&vT[cur][(i * 16 + m) * 32]), half);
      acc[i] = __builtin_amdgcn_wmma_f32_16x16x32_bf16(false, ap.v, false, bv.v,
                                                       (short)0, acc[i], false, false);
    }

#if HAS_TDM
    if (wid == 0) __builtin_amdgcn_s_wait_tensorcnt(0);  // next K tile landed
#endif
    __syncthreads();  // buffer swap: readers of `cur` done, `nxt` complete
  }

  // normalize and emit enc (bf16) laid out (B*T, NH*H)
#pragma unroll
  for (int r = 0; r < 8; ++r) rowsum[r] = 1.0f / rowsum[r];
#pragma unroll
  for (int i = 0; i < 16; ++i)
#pragma unroll
    for (int r = 0; r < 8; ++r) {
      size_t row = (size_t)(rowbase + half * 8 + r);
      enc[row * D_ + head * H_ + i * 16 + m] = f2bf(acc[i][r] * rowsum[r]);
    }
}

// ---------------- launch -------------------------------------------------------
extern "C" void kernel_launch(void* const* d_in, const int* in_sizes, int n_in,
                              void* d_out, int out_size, void* d_ws, size_t ws_size,
                              hipStream_t stream) {
  (void)in_sizes; (void)n_in; (void)out_size; (void)ws_size;
  const float* x   = (const float*)d_in[0];
  const float* mem = (const float*)d_in[1];
  const unsigned char* mmask = (const unsigned char*)d_in[2];
  const float* rs  = (const float*)d_in[3];
  const float* qw  = (const float*)d_in[4];
  const float* kvw = (const float*)d_in[5];
  const float* ow  = (const float*)d_in[6];

  char* w = (char*)d_ws;
  auto take = [&](size_t elems) { u16* p = (u16*)w; w += elems * 2; return p; };
  u16* xn  = take((size_t)4096 * 1024);   // rmsnorm(x) bf16
  u16* mn  = take((size_t)16384 * 1024);  // rmsnorm(mem) bf16
  u16* qwt = take((size_t)1024 * 1024);   // q_w  (NxK)
  u16* kwt = take((size_t)256 * 1024);    // k_w  (NxK)
  u16* vwt = take((size_t)256 * 1024);    // v_w  (NxK)
  u16* owt = take((size_t)1024 * 1024);   // out_w (NxK)
  u16* qb  = take((size_t)4096 * 1024);   // q  (B*T, NH*H)
  u16* kb  = take((size_t)16384 * 256);   // k  (B*S, H)
  u16* vb  = take((size_t)16384 * 256);   // v  (B*S, H)
  u16* enc = take((size_t)4096 * 1024);   // attention output

  conv_qw <<<4096, 256, 0, stream>>>(qw, qwt);
  conv_kvw<<<1024, 256, 0, stream>>>(kvw, kwt, vwt);
  conv_ow <<<4096, 256, 0, stream>>>(ow, owt);
  rmsnorm_bf16<<<4096, 256, 0, stream>>>(x, rs, xn);
  rmsnorm_bf16<<<16384, 256, 0, stream>>>(mem, rs, mn);

  gemm_wmma<true><<<2048, 256, 0, stream>>>(xn, qwt, qb, 4096, 1024, 1024);
  gemm_wmma<true><<<2048, 256, 0, stream>>>(mn, kwt, kb, 16384, 256, 1024);
  gemm_wmma<true><<<2048, 256, 0, stream>>>(mn, vwt, vb, 16384, 256, 1024);

  rope_bf16<<<8192, 256, 0, stream>>>(qb, 4, 511, 2048, 0.0625f, 4096 * 4 * 128);
  rope_bf16<<<8192, 256, 0, stream>>>(kb, 1, 2047, 0, 1.0f, 16384 * 128);

  attn_flash<<<128, 256, 0, stream>>>(qb, kb, vb, mmask, enc);

  gemm_wmma<false><<<2048, 256, 0, stream>>>(enc, owt, d_out, 4096, 1024, 1024);
}
